// DualTierMiras_14748917694645
// MI455X (gfx1250) — compile-verified, matrix operand order checked by reading, not run on hardware
//
#include <hip/hip_runtime.h>
#include <hip/hip_bf16.h>
#include <math.h>

#define SCALE_F 0.1f
#define LN_EPS_F 1e-5f

typedef __attribute__((ext_vector_type(16))) __bf16 v16bf;
typedef __attribute__((ext_vector_type(8)))  __bf16 v8bf;
typedef __attribute__((ext_vector_type(8)))  float  v8f;

// Load a 16-element bf16 WMMA fragment: per-lane two contiguous 16-byte chunks,
// elements 0..7 = K[kbase..kbase+7], elements 8..15 = K[kbase+16..kbase+23]
// (matches the 16-bit A/B VGPR layout tables in CDNA5 ISA).
static __device__ __forceinline__ v16bf load_frag(const __bf16* p) {
    v8bf lo = *reinterpret_cast<const v8bf*>(p);
    v8bf hi = *reinterpret_cast<const v8bf*>(p + 16);
    return __builtin_shufflevector(lo, hi, 0, 1, 2, 3, 4, 5, 6, 7,
                                           8, 9, 10, 11, 12, 13, 14, 15);
}

static __device__ __forceinline__ v8f zero_v8f() {
    v8f z;
#pragma unroll
    for (int e = 0; e < 8; ++e) z[e] = 0.0f;
    return z;
}

static __device__ __forceinline__ v8f wmma_bf16(v16bf a, v16bf b, v8f c) {
    return __builtin_amdgcn_wmma_f32_16x16x32_bf16(false, a, false, b, (short)0,
                                                   c, false, false);
}

// ---------------------------------------------------------------------------
// Kernel 1: per-row LayerNorm + bf16 conversion of k.  d == 2048 (8 x 256).
// ---------------------------------------------------------------------------
__global__ __launch_bounds__(256) void ln_prep_kernel(
    const float* __restrict__ k, const float* __restrict__ gamma,
    const float* __restrict__ beta, __bf16* __restrict__ kbf,
    __bf16* __restrict__ knbf, int d) {
    const int row = blockIdx.x;
    const int t = threadIdx.x;
    const float* kr = k + (size_t)row * d;
    float vals[8];
    float s = 0.0f, sq = 0.0f;
#pragma unroll
    for (int i = 0; i < 8; ++i) {
        float v = kr[t + i * 256];
        vals[i] = v;
        s += v;
        sq += v * v;
    }
    __shared__ float ssum[256];
    __shared__ float ssq[256];
    ssum[t] = s;
    ssq[t] = sq;
    __syncthreads();
    for (int off = 128; off > 0; off >>= 1) {
        if (t < off) {
            ssum[t] += ssum[t + off];
            ssq[t] += ssq[t + off];
        }
        __syncthreads();
    }
    const float inv_d = 1.0f / (float)d;
    const float mu = ssum[0] * inv_d;
    const float var = ssq[0] * inv_d - mu * mu;
    const float rstd = rsqrtf(var + LN_EPS_F);
#pragma unroll
    for (int i = 0; i < 8; ++i) {
        const int j = t + i * 256;
        kbf[(size_t)row * d + j] = (__bf16)vals[i];
        knbf[(size_t)row * d + j] =
            (__bf16)((vals[i] - mu) * rstd * gamma[j] + beta[j]);
    }
}

// ---------------------------------------------------------------------------
// Kernel 2: build both tier matrices in bf16, stored as W[col][k] row-major
// (exactly the layout the WMMA B-fragment loader wants).
// W[j][i] = SCALE*tanh(dot(B[j,:r], C[i,:r])) + (i==j)*D[i]
// ---------------------------------------------------------------------------
__global__ __launch_bounds__(256) void build_w_kernel(
    const float* __restrict__ Bf, const float* __restrict__ Cf,
    const float* __restrict__ Df, const float* __restrict__ Bd,
    const float* __restrict__ Cd, const float* __restrict__ Dd,
    __bf16* __restrict__ wf, __bf16* __restrict__ wd, int d, int r) {
    const int j = blockIdx.y;
    const int i = blockIdx.x * blockDim.x + threadIdx.x;
    float af = 0.0f, ad = 0.0f;
    for (int rr = 0; rr < r; ++rr) {
        af += Bf[(size_t)j * r + rr] * Cf[(size_t)i * r + rr];
        ad += Bd[(size_t)j * r + rr] * Cd[(size_t)i * r + rr];
    }
    float vf = SCALE_F * tanhf(af);
    float vd = SCALE_F * tanhf(ad);
    if (i == j) {
        vf += Df[i];
        vd += Dd[i];
    }
    wf[(size_t)j * d + i] = (__bf16)vf;
    wd[(size_t)j * d + i] = (__bf16)vd;
}

// ---------------------------------------------------------------------------
// Kernel 3: fp32 -> bf16 elementwise convert (gate_W1 is already [out][in]).
// ---------------------------------------------------------------------------
__global__ void cvt_bf16_kernel(const float* __restrict__ src,
                                __bf16* __restrict__ dst, size_t n) {
    size_t idx = (size_t)blockIdx.x * blockDim.x + threadIdx.x;
    const size_t stride = (size_t)gridDim.x * blockDim.x;
    for (; idx < n; idx += stride) dst[idx] = (__bf16)src[idx];
}

// ---------------------------------------------------------------------------
// Kernel 4: gate GEMM  H = silu(A @ B^T + bias), A=[M,K] bf16, B=[N,K] bf16.
// Block tile 128x256, 8 waves in 2(M)x4(N), 64x64 per wave (4x4 fragments),
// K-step 32.  16 WMMAs per 16 b128 loads per K-step.
// ---------------------------------------------------------------------------
__global__ __launch_bounds__(256) void gate_gemm_kernel(
    const __bf16* __restrict__ A, const __bf16* __restrict__ B,
    const float* __restrict__ bias, __bf16* __restrict__ H, int M, int N,
    int K) {
    const int lane = threadIdx.x & 31;
    const int wave = threadIdx.x >> 5;
    const int wm = wave >> 2;        // 0..1
    const int wn = wave & 3;         // 0..3
    const int mbase = blockIdx.y * 128 + wm * 64;
    const int nbase = blockIdx.x * 256 + wn * 64;
    const int lrow = lane & 15;
    const int lk = (lane >> 4) * 8;

    v8f acc[4][4];
#pragma unroll
    for (int i = 0; i < 4; ++i)
#pragma unroll
        for (int j = 0; j < 4; ++j) acc[i][j] = zero_v8f();

    const __bf16* ap[4];
    const __bf16* bp[4];
#pragma unroll
    for (int i = 0; i < 4; ++i) {
        ap[i] = A + (size_t)(mbase + i * 16 + lrow) * K + lk;
        bp[i] = B + (size_t)(nbase + i * 16 + lrow) * K + lk;
    }

    for (int kk = 0; kk < K; kk += 32) {
        v16bf a[4], b[4];
#pragma unroll
        for (int i = 0; i < 4; ++i) a[i] = load_frag(ap[i] + kk);
#pragma unroll
        for (int j = 0; j < 4; ++j) b[j] = load_frag(bp[j] + kk);
        __builtin_prefetch(ap[0] + kk + 128, 0, 3);
        __builtin_prefetch(ap[2] + kk + 128, 0, 3);
#pragma unroll
        for (int i = 0; i < 4; ++i)
#pragma unroll
            for (int j = 0; j < 4; ++j) acc[i][j] = wmma_bf16(a[i], b[j], acc[i][j]);
    }

#pragma unroll
    for (int i = 0; i < 4; ++i) {
#pragma unroll
        for (int j = 0; j < 4; ++j) {
            const int col = nbase + j * 16 + (lane & 15);
            const float bcol = bias[col];
#pragma unroll
            for (int v = 0; v < 8; ++v) {
                const int rowm = mbase + i * 16 + (lane >> 4) * 8 + v;
                const float x = acc[i][j][v] + bcol;
                const float sg = 1.0f / (1.0f + __expf(-x));
                H[(size_t)rowm * N + col] = (__bf16)(x * sg);
            }
        }
    }
}

// ---------------------------------------------------------------------------
// Kernel 5: logit[n] = dot(H[n,:], W2) + b2 + base; wfast[n] = sigmoid.
// Deterministic LDS tree reduction.
// ---------------------------------------------------------------------------
__global__ __launch_bounds__(256) void logit_kernel(
    const __bf16* __restrict__ H, const float* __restrict__ W2,
    const float* __restrict__ b2, const float* __restrict__ base,
    float* __restrict__ wfast, int d) {
    const int row = blockIdx.x;
    const int t = threadIdx.x;
    const __bf16* hr = H + (size_t)row * d;
    float s = 0.0f;
    for (int j = t; j < d; j += 256) s += (float)hr[j] * W2[j];
    __shared__ float red[256];
    red[t] = s;
    __syncthreads();
    for (int off = 128; off > 0; off >>= 1) {
        if (t < off) red[t] += red[t + off];
        __syncthreads();
    }
    if (t == 0) {
        const float logit = red[0] + b2[0] + base[0];
        wfast[row] = 1.0f / (1.0f + __expf(-logit));
    }
}

// ---------------------------------------------------------------------------
// Kernel 6: dual-tier GEMM with fused blend epilogue.
//   out[n][j] = w[n]*(k @ Wf^T)[n][j] + (1-w[n])*(k @ Wd^T)[n][j]
// Block tile 128x128, 8 waves in 2(M)x4(N), wave tile 64x32 per tier
// (4 M-frags x 2 N-frags x 2 tiers): 16 WMMAs per 16 b128 loads per K-step;
// A fragments shared across both tiers.
// ---------------------------------------------------------------------------
__global__ __launch_bounds__(256) void tier_gemm_kernel(
    const __bf16* __restrict__ A, const __bf16* __restrict__ BF,
    const __bf16* __restrict__ BD, const float* __restrict__ wgate,
    float* __restrict__ out, int M, int N, int K) {
    const int lane = threadIdx.x & 31;
    const int wave = threadIdx.x >> 5;
    const int wm = wave >> 2;        // 0..1
    const int wn = wave & 3;         // 0..3
    const int mbase = blockIdx.y * 128 + wm * 64;
    const int nbase = blockIdx.x * 128 + wn * 32;
    const int lrow = lane & 15;
    const int lk = (lane >> 4) * 8;

    v8f accf[4][2], accd[4][2];
#pragma unroll
    for (int i = 0; i < 4; ++i)
#pragma unroll
        for (int j = 0; j < 2; ++j) {
            accf[i][j] = zero_v8f();
            accd[i][j] = zero_v8f();
        }

    const __bf16* ap[4];
#pragma unroll
    for (int i = 0; i < 4; ++i)
        ap[i] = A + (size_t)(mbase + i * 16 + lrow) * K + lk;
    const __bf16* bfp[2];
    const __bf16* bdp[2];
#pragma unroll
    for (int j = 0; j < 2; ++j) {
        bfp[j] = BF + (size_t)(nbase + j * 16 + lrow) * K + lk;
        bdp[j] = BD + (size_t)(nbase + j * 16 + lrow) * K + lk;
    }

    for (int kk = 0; kk < K; kk += 32) {
        v16bf a[4], f[2], dd[2];
#pragma unroll
        for (int i = 0; i < 4; ++i) a[i] = load_frag(ap[i] + kk);
#pragma unroll
        for (int j = 0; j < 2; ++j) {
            f[j] = load_frag(bfp[j] + kk);
            dd[j] = load_frag(bdp[j] + kk);
        }
        __builtin_prefetch(ap[0] + kk + 128, 0, 3);
        __builtin_prefetch(ap[2] + kk + 128, 0, 3);
#pragma unroll
        for (int i = 0; i < 4; ++i)
#pragma unroll
            for (int j = 0; j < 2; ++j) {
                accf[i][j] = wmma_bf16(a[i], f[j], accf[i][j]);
                accd[i][j] = wmma_bf16(a[i], dd[j], accd[i][j]);
            }
    }

#pragma unroll
    for (int i = 0; i < 4; ++i) {
#pragma unroll
        for (int j = 0; j < 2; ++j) {
            const int col = nbase + j * 16 + (lane & 15);
#pragma unroll
            for (int v = 0; v < 8; ++v) {
                const int rowm = mbase + i * 16 + (lane >> 4) * 8 + v;
                const float w = wgate[rowm];
                out[(size_t)rowm * N + col] =
                    w * accf[i][j][v] + (1.0f - w) * accd[i][j][v];
            }
        }
    }
}

// ---------------------------------------------------------------------------
// Host-side launch: carve workspace, run the 6-kernel pipeline on `stream`.
// ---------------------------------------------------------------------------
extern "C" void kernel_launch(void* const* d_in, const int* in_sizes, int n_in,
                              void* d_out, int out_size, void* d_ws,
                              size_t ws_size, hipStream_t stream) {
    const float* k      = (const float*)d_in[0];
    const float* fast_B = (const float*)d_in[1];
    const float* fast_C = (const float*)d_in[2];
    const float* fast_D = (const float*)d_in[3];
    const float* deep_B = (const float*)d_in[4];
    const float* deep_C = (const float*)d_in[5];
    const float* deep_D = (const float*)d_in[6];
    const float* ln_g   = (const float*)d_in[7];
    const float* ln_b   = (const float*)d_in[8];
    const float* gW1    = (const float*)d_in[9];
    const float* gb1    = (const float*)d_in[10];
    const float* gW2    = (const float*)d_in[11];
    const float* gb2    = (const float*)d_in[12];
    const float* base   = (const float*)d_in[13];

    const int d = in_sizes[7];             // 2048
    const int N = in_sizes[0] / d;         // 8192
    const int r = in_sizes[1] / d;         // 32

    char* ws = (char*)d_ws;
    size_t off = 0;
    auto carve = [&](size_t bytes) -> char* {
        char* p = ws + off;
        off += (bytes + 255) & ~(size_t)255;
        return p;
    };
    __bf16* kbf  = (__bf16*)carve((size_t)N * d * sizeof(__bf16));
    __bf16* knbf = (__bf16*)carve((size_t)N * d * sizeof(__bf16));
    __bf16* wfb  = (__bf16*)carve((size_t)d * d * sizeof(__bf16));
    __bf16* wdb  = (__bf16*)carve((size_t)d * d * sizeof(__bf16));
    __bf16* w1b  = (__bf16*)carve((size_t)d * d * sizeof(__bf16));
    __bf16* hbf  = (__bf16*)carve((size_t)N * d * sizeof(__bf16));
    float*  wg   = (float*)carve((size_t)N * sizeof(float));

    // 1) LayerNorm + bf16 conversion of k
    ln_prep_kernel<<<dim3(N), dim3(256), 0, stream>>>(k, ln_g, ln_b, kbf, knbf,
                                                      d);
    // 2) Build both tier matrices in bf16 (B-operand layout)
    build_w_kernel<<<dim3(d / 256, d), dim3(256), 0, stream>>>(
        fast_B, fast_C, fast_D, deep_B, deep_C, deep_D, wfb, wdb, d, r);
    // 3) gate_W1 -> bf16
    cvt_bf16_kernel<<<dim3(2048), dim3(256), 0, stream>>>(gW1, w1b,
                                                          (size_t)d * d);
    // 4) h = silu(LN(k) @ W1^T + b1)   block tile 128x256
    gate_gemm_kernel<<<dim3(d / 256, N / 128), dim3(256), 0, stream>>>(
        knbf, w1b, gb1, hbf, N, d, d);
    // 5) w_fast = sigmoid(h @ W2^T + b2 + base)
    logit_kernel<<<dim3(N), dim3(256), 0, stream>>>(hbf, gW2, gb2, base, wg, d);
    // 6) out = w*v_fast + (1-w)*v_deep (dual-tier GEMM, fused blend)
    //    block tile 128x128
    tier_gemm_kernel<<<dim3(d / 128, N / 128), dim3(256), 0, stream>>>(
        kbf, wfb, wdb, wg, (float*)d_out, N, d, d);

    (void)n_in;
    (void)out_size;
    (void)ws_size;
}
